// SchNetBackbone_7687991460554
// MI455X (gfx1250) — matrix operand order repaired and checked by reading, not compiled, not run on hardware
//
#include <hip/hip_runtime.h>
#include <hip/hip_bf16.h>

#define NN 10000
#define NPAD 10112            // 79 * 128, padded node-row count (no guards in GEMMs)
#define EE 320000
#define HIDC 128
#define NFC 128
#define NGC 50
#define NBC 6
#define CUTOFF 10.0f

typedef __attribute__((ext_vector_type(16))) _Float16 v16h;
typedef __attribute__((ext_vector_type(8)))  float    v8f;

union FragU { v16h v; uint4 u[2]; };

// A fragment (16x32 f16, M x K). Lane l: row = l&15, K-halves per ISA layout:
// contiguous chunks [8*kh, 8*kh+8) and [16+8*kh, 16+8*kh+8) relative to k0.
__device__ __forceinline__ v16h ld_a_frag(const _Float16* As, int lda, int row0,
                                          int k0, int lane) {
  int r = lane & 15, kh = lane >> 4;
  const _Float16* p = As + (row0 + r) * lda + k0 + 8 * kh;
  FragU f;
  f.u[0] = *(const uint4*)(p);
  f.u[1] = *(const uint4*)(p + 16);
  return f.v;
}

// B fragment (32x16 f16, K x N) from K-contiguous (transposed) storage BsT[n][k].
// Lane l: col n = l&15, holds K chunks [k0+8*kh, k0+8*kh+8) and [k0+16+8*kh, k0+16+8*kh+8).
__device__ __forceinline__ v16h ld_b_frag(const _Float16* BsT, int ldk, int n0,
                                          int k0, int lane) {
  int n = lane & 15, kh = lane >> 4;
  const _Float16* p = BsT + (n0 + n) * ldk + k0 + 8 * kh;
  FragU f;
  f.u[0] = *(const uint4*)(p);
  f.u[1] = *(const uint4*)(p + 16);
  return f.v;
}

// shifted softplus via HW transcendentals (v_exp_f32 / v_log_f32):
// ssp(x) = max(x,0) + log(1 + exp(-|x|)) - ln2   (stable for all x)
__device__ __forceinline__ float sspf(float v) {
  float t = __expf(-fabsf(v));
  return fmaxf(v, 0.f) + __logf(1.f + t) - 0.69314718055994530942f;
}

// ---------------- prologue kernels ----------------

__global__ void k_edge_geom(const float* __restrict__ pos, const int* __restrict__ ei,
                            float* __restrict__ dist, float* __restrict__ cutv) {
  int e = blockIdx.x * blockDim.x + threadIdx.x;
  if (e >= EE) return;
  int s = ei[e], t = ei[EE + e];
  float dx = pos[3 * s + 0] - pos[3 * t + 0];
  float dy = pos[3 * s + 1] - pos[3 * t + 1];
  float dz = pos[3 * s + 2] - pos[3 * t + 2];
  float d = sqrtf(dx * dx + dy * dy + dz * dz);
  dist[e] = d;
  cutv[e] = 0.5f * (__cosf(d * (3.14159265358979323846f / CUTOFF)) + 1.0f);
}

// h[0:NN) = emb[z]; h[NN:NPAD) = 0   (pad rows stay self-contained)
__global__ void k_embed(const int* __restrict__ z, const float* __restrict__ emb,
                        float* __restrict__ h) {
  int idx = blockIdx.x * blockDim.x + threadIdx.x;
  if (idx >= NPAD * HIDC) return;
  int i = idx >> 7, c = idx & 127;
  h[idx] = (i < NN) ? emb[z[i] * HIDC + c] : 0.f;
}

__global__ void k_zero(float* __restrict__ p, int nElem) {
  int idx = blockIdx.x * blockDim.x + threadIdx.x;
  if (idx < nElem) p[idx] = 0.f;
}

// src f32 [K][Nc] row-major  ->  dst f16 [Nc][Kpad] (K zero-padded to Kpad)
__global__ void k_convT(const float* __restrict__ src, _Float16* __restrict__ dst,
                        int K, int Nc, int Kpad) {
  int idx = blockIdx.x * blockDim.x + threadIdx.x;
  if (idx >= Nc * Kpad) return;
  int n = idx / Kpad, k = idx - n * Kpad;
  dst[idx] = (k < K) ? (_Float16)src[k * Nc + n] : (_Float16)0.f;
}

// ---------------- generic WMMA GEMM (row-padded, guard-free, compile-time epilogue):
// C = op(A[NPAD,128] @ B[128,ncols] + bias) (+resid)

template <bool HASBIAS, bool DOSSP, bool HASRESID>
__global__ __launch_bounds__(256) void k_gemm128(
    const float* __restrict__ A, const _Float16* __restrict__ BT,
    const float* __restrict__ bias, const float* __restrict__ resid,
    float* __restrict__ Cout, int ncols) {
  extern __shared__ char smem[];
  _Float16* As = (_Float16*)smem;        // 128*128 f16
  _Float16* Bs = As + 128 * 128;         // ncols*128 f16
  int t = threadIdx.x;
  int base = blockIdx.x * 128;
  for (int idx = t; idx < 128 * 128; idx += 256)
    As[idx] = (_Float16)A[base * 128 + idx];
  for (int idx = t; idx < ncols * 128; idx += 256) Bs[idx] = BT[idx];
  __syncthreads();

  int lane = t & 31, wv = t >> 5;
  int row0 = wv * 16;
  int n = lane & 15, kh = lane >> 4;
  int ntiles = ncols >> 4;
  for (int nt = 0; nt < ntiles; ++nt) {
    v8f acc = {};
#pragma unroll
    for (int k0 = 0; k0 < 128; k0 += 32) {
      v16h a = ld_a_frag(As, 128, row0, k0, lane);
      v16h b = ld_b_frag(Bs, 128, nt * 16, k0, lane);
      acc = __builtin_amdgcn_wmma_f32_16x16x32_f16(false, a, false, b,
                                                   (short)0, acc, false, false);
    }
    int gcol = nt * 16 + n;
    float bv = HASBIAS ? bias[gcol] : 0.f;
#pragma unroll
    for (int r = 0; r < 8; ++r) {
      int grow = base + row0 + r + 8 * kh;
      float v = acc[r] + bv;
      if (DOSSP) v = sspf(v);
      if (HASRESID) v += resid[grow * 128 + gcol];
      Cout[grow * 128 + gcol] = v;
    }
  }
}

// ---------------- fused edge kernel: RBF -> filter MLP (2 WMMA GEMMs) -> cutoff
//                  -> gather x[src] -> scatter-add agg[dst] (hw f32 atomics)

__global__ __launch_bounds__(256) void k_edge_msg(
    const float* __restrict__ dist, const float* __restrict__ cutv,
    const int* __restrict__ ei,
    const _Float16* __restrict__ w1T, const _Float16* __restrict__ w2T,
    const float* __restrict__ b1, const float* __restrict__ b2,
    const float* __restrict__ x, float* __restrict__ agg) {
  extern __shared__ char smem[];
  _Float16* Ea  = (_Float16*)smem;       // 128 x 64 (edge_attr tile, padded)
  _Float16* Bs1 = Ea + 128 * 64;         // w1T [128][64]
  _Float16* Tm  = Bs1 + 128 * 64;        // intermediate 128 x 128
  _Float16* Bs2 = Tm + 128 * 128;        // w2T [128][128]
  float* Ds = (float*)(Bs2 + 128 * 128); // d per edge
  float* Cs = Ds + 128;                  // cutoff per edge
  int* srcs = (int*)(Cs + 128);
  int* dsts = srcs + 128;

  int t = threadIdx.x;
  int ebase = blockIdx.x * 128;          // EE is an exact multiple of 128
  if (t < 128) {
    int e = ebase + t;
    Ds[t] = dist[e];
    Cs[t] = cutv[e];
    srcs[t] = ei[e];
    dsts[t] = ei[EE + e];
  }
  __syncthreads();

  const float spacing = CUTOFF / (float)(NGC - 1);
  const float coeff = -0.5f / (spacing * spacing);
  for (int idx = t; idx < 128 * 64; idx += 256) {
    int i = idx >> 6, g = idx & 63;
    float v = 0.f;
    if (g < NGC) {
      float diff = Ds[i] - (float)g * spacing;
      v = __expf(coeff * diff * diff);
    }
    Ea[idx] = (_Float16)v;
    Bs1[idx] = w1T[idx];
  }
  for (int idx = t; idx < 128 * 128; idx += 256) Bs2[idx] = w2T[idx];
  __syncthreads();

  int lane = t & 31, wv = t >> 5;
  int row0 = wv * 16;
  int n = lane & 15, kh = lane >> 4;

  // phase 1: Tm = ssp(Ea @ w1 + b1)   (K = 64)
  for (int nt = 0; nt < 8; ++nt) {
    v8f acc = {};
#pragma unroll
    for (int k0 = 0; k0 < 64; k0 += 32) {
      v16h a = ld_a_frag(Ea, 64, row0, k0, lane);
      v16h b = ld_b_frag(Bs1, 64, nt * 16, k0, lane);
      acc = __builtin_amdgcn_wmma_f32_16x16x32_f16(false, a, false, b,
                                                   (short)0, acc, false, false);
    }
    int gcol = nt * 16 + n;
    float b1v = b1[gcol];
#pragma unroll
    for (int r = 0; r < 8; ++r) {
      int i = row0 + r + 8 * kh;
      Tm[i * 128 + gcol] = (_Float16)sspf(acc[r] + b1v);
    }
  }
  __syncthreads();

  // phase 2: W = (Tm @ w2 + b2) * C ; msg = W * x[src] ; agg[dst] += msg
  for (int nt = 0; nt < 8; ++nt) {
    v8f acc = {};
#pragma unroll
    for (int k0 = 0; k0 < 128; k0 += 32) {
      v16h a = ld_a_frag(Tm, 128, row0, k0, lane);
      v16h b = ld_b_frag(Bs2, 128, nt * 16, k0, lane);
      acc = __builtin_amdgcn_wmma_f32_16x16x32_f16(false, a, false, b,
                                                   (short)0, acc, false, false);
    }
    int gcol = nt * 16 + n;
    float b2v = b2[gcol];
#pragma unroll
    for (int r = 0; r < 8; ++r) {
      int i = row0 + r + 8 * kh;
      float wfil = (acc[r] + b2v) * Cs[i];
      float msg = wfil * x[srcs[i] * 128 + gcol];
      unsafeAtomicAdd(&agg[dsts[i] * 128 + gcol], msg);
    }
  }
}

// ---------------- epilogue: out = ssp_row @ out_w2 + out_b2

__global__ void k_outdot(const float* __restrict__ tbuf, const float* __restrict__ w2,
                         const float* __restrict__ b2, float* __restrict__ out) {
  int i = blockIdx.x * blockDim.x + threadIdx.x;
  if (i >= NN) return;
  float s = b2[0];
  const float* r = tbuf + i * 128;  // cols 0..63 valid
  for (int j = 0; j < 64; ++j) s += r[j] * w2[j];
  out[i] = s;
}

// ---------------- launch ----------------

extern "C" void kernel_launch(void* const* d_in, const int* in_sizes, int n_in,
                              void* d_out, int out_size, void* d_ws, size_t ws_size,
                              hipStream_t stream) {
  const int*   z    = (const int*)  d_in[0];
  const float* pos  = (const float*)d_in[1];
  const int*   ei   = (const int*)  d_in[2];
  const float* emb  = (const float*)d_in[3];
  const float* mw1  = (const float*)d_in[4];
  const float* mb1  = (const float*)d_in[5];
  const float* mw2  = (const float*)d_in[6];
  const float* mb2  = (const float*)d_in[7];
  const float* cl1  = (const float*)d_in[8];
  const float* cl2  = (const float*)d_in[9];
  const float* cl2b = (const float*)d_in[10];
  const float* lin  = (const float*)d_in[11];
  const float* linb = (const float*)d_in[12];
  const float* ow1  = (const float*)d_in[13];
  const float* ob1  = (const float*)d_in[14];
  const float* ow2  = (const float*)d_in[15];
  const float* ob2  = (const float*)d_in[16];
  float* out = (float*)d_out;

  // workspace layout (~19.5 MB total), node buffers row-padded to NPAD
  float* h    = (float*)d_ws;          // NPAD*128
  float* x    = h + NPAD * HIDC;       // NPAD*128
  float* agg  = x + NPAD * HIDC;       // NPAD*128
  float* dist = agg + NPAD * HIDC;     // E
  float* cutv = dist + EE;             // E
  _Float16* w1T  = (_Float16*)(cutv + EE);     // NB * 128*64
  _Float16* w2T  = w1T + NBC * 128 * 64;       // NB * 128*128
  _Float16* cl1T = w2T + NBC * 128 * 128;
  _Float16* cl2T = cl1T + NBC * 128 * 128;
  _Float16* linT = cl2T + NBC * 128 * 128;
  _Float16* ow1T = linT + NBC * 128 * 128;     // 64*128

  // per-call f16 transposed weight conversion
  for (int k = 0; k < NBC; ++k) {
    k_convT<<<32, 256, 0, stream>>>(mw1 + k * NGC * NFC, w1T + k * 128 * 64, NGC, NFC, 64);
    k_convT<<<64, 256, 0, stream>>>(mw2 + k * NFC * NFC, w2T + k * 128 * 128, NFC, NFC, 128);
    k_convT<<<64, 256, 0, stream>>>(cl1 + k * HIDC * NFC, cl1T + k * 128 * 128, HIDC, NFC, 128);
    k_convT<<<64, 256, 0, stream>>>(cl2 + k * NFC * HIDC, cl2T + k * 128 * 128, NFC, HIDC, 128);
    k_convT<<<64, 256, 0, stream>>>(lin + k * HIDC * HIDC, linT + k * 128 * 128, HIDC, HIDC, 128);
  }
  k_convT<<<32, 256, 0, stream>>>(ow1, ow1T, HIDC, 64, 128);

  k_edge_geom<<<EE / 256, 256, 0, stream>>>(pos, ei, dist, cutv);
  k_embed<<<(NPAD * HIDC) / 256, 256, 0, stream>>>(z, emb, h);

  const int gemmGrid = NPAD / 128;                                   // 79
  const size_t gemmLds = (size_t)2 * 128 * 128 * sizeof(_Float16);   // 64 KB
  const size_t edgeLds = (size_t)(2 * 128 * 64 + 2 * 128 * 128) * sizeof(_Float16)
                       + 2 * 128 * sizeof(float) + 2 * 128 * sizeof(int); // ~100 KB

  for (int k = 0; k < NBC; ++k) {
    // x = h @ cl1_w[k]
    k_gemm128<false, false, false><<<gemmGrid, 256, gemmLds, stream>>>(
        h, cl1T + k * 128 * 128, nullptr, nullptr, x, 128);
    k_zero<<<(NPAD * HIDC) / 256, 256, 0, stream>>>(agg, NPAD * HIDC);
    // fused filter MLP + gather/scatter message passing
    k_edge_msg<<<EE / 128, 256, edgeLds, stream>>>(dist, cutv, ei,
                                                   w1T + k * 128 * 64,
                                                   w2T + k * 128 * 128,
                                                   mb1 + k * NFC, mb2 + k * NFC,
                                                   x, agg);
    // x = ssp(agg @ cl2_w + cl2_b)
    k_gemm128<true, true, false><<<gemmGrid, 256, gemmLds, stream>>>(
        agg, cl2T + k * 128 * 128, cl2b + k * HIDC, nullptr, x, 128);
    // h = h + x @ lin_w + lin_b
    k_gemm128<true, false, true><<<gemmGrid, 256, gemmLds, stream>>>(
        x, linT + k * 128 * 128, linb + k * HIDC, h, h, 128);
  }

  // x[:, :64] = ssp(h @ out_w1 + out_b1) ; out = x[:, :64] @ out_w2 + out_b2
  k_gemm128<true, true, false><<<gemmGrid, 256, gemmLds, stream>>>(
      h, ow1T, ob1, nullptr, x, 64);
  k_outdot<<<(NN + 255) / 256, 256, 0, stream>>>(x, ow2, ob2, out);
}
